// HunyuanMoE_10763188044205
// MI455X (gfx1250) — matrix operand order repaired and checked by reading, not compiled, not run on hardware
//
#include <hip/hip_runtime.h>
#include <hip/hip_bf16.h>

#define TOKENS 4096
#define HID    2048
#define INTER  1024
#define NEXP   16

typedef __attribute__((ext_vector_type(16))) __bf16 v16bf;
typedef __attribute__((ext_vector_type(8)))  float  v8f;

union BF16Frag { v16bf v; uint4 q[2]; };

#if defined(__gfx1250__) && __has_builtin(__builtin_amdgcn_global_load_async_to_lds_b128)
#define HAVE_ASYNC_LDS 1
#else
#define HAVE_ASYNC_LDS 0
#endif

#if HAVE_ASYNC_LDS
typedef int b128vec __attribute__((vector_size(4 * sizeof(int))));
#if __has_builtin(__builtin_amdgcn_s_wait_asynccnt)
#define WAIT_ASYNC() __builtin_amdgcn_s_wait_asynccnt(0)
#else
#define WAIT_ASYNC() asm volatile("s_wait_asynccnt 0" ::: "memory")
#endif
#define ASYNC_LDS_B128(gsrc, ldst)                                             \
    __builtin_amdgcn_global_load_async_to_lds_b128(                            \
        (__attribute__((address_space(1))) b128vec*)(gsrc),                    \
        (__attribute__((address_space(3))) b128vec*)(ldst), 0, 0)
#endif

static __device__ __forceinline__ unsigned short f32_to_bf16(float a) {
    unsigned int u = __float_as_uint(a);
    u += 0x7FFFu + ((u >> 16) & 1u);       // round-to-nearest-even
    return (unsigned short)(u >> 16);
}
static __device__ __forceinline__ unsigned int pack2_bf16(float a, float b) {
    unsigned int ua = __float_as_uint(a);
    unsigned int ub = __float_as_uint(b);
    ua += 0x7FFFu + ((ua >> 16) & 1u);
    ub += 0x7FFFu + ((ub >> 16) & 1u);
    return (ua >> 16) | (ub & 0xFFFF0000u);
}
static __device__ __forceinline__ float fast_sigmoid(float x) {
    // 1/(1+exp(-x)) with hardware v_rcp_f32 (bf16-quantized output -> plenty of precision)
    return __builtin_amdgcn_rcpf(1.0f + __expf(-x));
}

// ---------------------------------------------------------------- x -> bf16
__global__ void cvt_x_kernel(const float* __restrict__ x, uint4* __restrict__ xb) {
    int i = blockIdx.x * blockDim.x + threadIdx.x;       // handles 8 floats
    const float4* src = (const float4*)x;
    float4 f0 = src[2 * i + 0];
    float4 f1 = src[2 * i + 1];
    uint4 o;
    o.x = pack2_bf16(f0.x, f0.y); o.y = pack2_bf16(f0.z, f0.w);
    o.z = pack2_bf16(f1.x, f1.y); o.w = pack2_bf16(f1.z, f1.w);
    xb[i] = o;
}

// ---------------------------------------------------------------- router
__global__ void router_kernel(const float* __restrict__ x, const float* __restrict__ gw,
                              int* __restrict__ cnt, int* __restrict__ pick_e,
                              float* __restrict__ pick_w) {
    int t = blockIdx.x * blockDim.x + threadIdx.x;
    if (t >= TOKENS) return;
    const float4* xr = (const float4*)(x + (size_t)t * HID);
    float acc[NEXP];
#pragma unroll
    for (int e = 0; e < NEXP; e++) acc[e] = 0.f;
    for (int i = 0; i < HID / 4; i++) {
        float4 xv = xr[i];
#pragma unroll
        for (int e = 0; e < NEXP; e++) {
            float4 wv = ((const float4*)(gw + (size_t)e * HID))[i];
            acc[e] += xv.x * wv.x + xv.y * wv.y + xv.z * wv.z + xv.w * wv.w;
        }
    }
    int e0 = 0; float l0 = acc[0];
#pragma unroll
    for (int e = 1; e < NEXP; e++) if (acc[e] > l0) { l0 = acc[e]; e0 = e; }
    int e1 = (e0 == 0) ? 1 : 0; float l1 = acc[e1];
#pragma unroll
    for (int e = 0; e < NEXP; e++) if (e != e0 && acc[e] > l1) { l1 = acc[e]; e1 = e; }
    // renormalized top-2 softmax == two-way softmax (denominator cancels)
    float w0 = fast_sigmoid(l0 - l1);
    float w1 = 1.0f - w0;
    pick_e[2 * t] = e0;  pick_e[2 * t + 1] = e1;
    pick_w[2 * t] = w0;  pick_w[2 * t + 1] = w1;
    atomicAdd(&cnt[e0], 1);
    atomicAdd(&cnt[e1], 1);
}

__global__ void scan_kernel(const int* __restrict__ cnt, int* __restrict__ base) {
    if (threadIdx.x == 0) {
        int run = 0;
        for (int e = 0; e < NEXP; e++) { base[e] = run; run += cnt[e]; }
    }
}

__global__ void assign_kernel(const int* __restrict__ pick_e, const float* __restrict__ pick_w,
                              const int* __restrict__ base, int* __restrict__ cursor,
                              int* __restrict__ tok_of, float* __restrict__ coef_of) {
    int t = blockIdx.x * blockDim.x + threadIdx.x;
    if (t >= TOKENS) return;
#pragma unroll
    for (int r = 0; r < 2; r++) {
        int e = pick_e[2 * t + r];
        int s = atomicAdd(&cursor[e], 1);
        int slot = base[e] + s;
        tok_of[slot] = t;
        coef_of[slot] = pick_w[2 * t + r];
    }
}

// ---------------------------------------------------------------- GEMM1: gate_up + SiLU*mul -> act (bf16)
// One kernel for shared (routed=0) and routed (routed=1) experts (dims coincide).
__global__ __launch_bounds__(256) void gemm1_kernel(
    const unsigned short* __restrict__ xb,   // [TOKENS][HID] bf16
    const float* __restrict__ w,             // [E][2I][H] or [2I][H]
    const float* __restrict__ bias,          // [E][2I] or [2I]
    const int* __restrict__ base, const int* __restrict__ cnt,
    const int* __restrict__ tok_of,
    unsigned short* __restrict__ act,        // [rows][INTER] bf16
    int routed) {
    constexpr int BM = 128, BN = 64, BK = 32, LDA = 40, LDB = 40;
    __shared__ __align__(16) unsigned short As[BM * LDA];
    __shared__ __align__(16) unsigned short Bg[BN * LDB];
    __shared__ __align__(16) unsigned short Bu[BN * LDB];

    const int e      = blockIdx.z;
    const int n_rows = routed ? cnt[e] : TOKENS;
    const int row0   = blockIdx.y * BM;
    if (row0 >= n_rows) return;
    const int sbase  = routed ? base[e] : 0;
    const float* we  = w    + (routed ? (size_t)e * (2 * INTER) * HID : 0);
    const float* be  = bias + (routed ? e * (2 * INTER) : 0);
    const int ncol0  = blockIdx.x * BN;     // gate column base in [0, INTER)

    const int tid  = threadIdx.x;
    const int lane = tid & 31, wid = tid >> 5;
    const int lr   = lane & 15, hi = lane >> 4;
    const int Moff = (wid & 3) * 32;
    const int Noff = (wid >> 2) * 32;

    // A loader: 2 threads per row, 16 bf16 each
    const int ar = tid >> 1;
    const int acol = (tid & 1) * 16;
    int arow = row0 + ar;
    int tokA;
    if (routed) tokA = tok_of[sbase + (arow < n_rows ? arow : 0)];
    else        tokA = arow;
    const unsigned short* aSrcBase = xb + (size_t)tokA * HID + acol;
    unsigned short* aDst0 = &As[ar * LDA + acol];
    unsigned short* aDst1 = &As[ar * LDA + acol + 8];

    // B loader: 4 threads per row, 8 fp32 each (convert to bf16)
    const int brr = tid >> 2;
    const int bcol = (tid & 3) * 8;
    const float* bgSrcBase = we + (size_t)(ncol0 + brr) * HID + bcol;
    const float* buSrcBase = we + (size_t)(INTER + ncol0 + brr) * HID + bcol;

    v8f accg[2][2], accu[2][2];
#pragma unroll
    for (int mi = 0; mi < 2; mi++)
#pragma unroll
        for (int ni = 0; ni < 2; ni++)
#pragma unroll
            for (int j = 0; j < 8; j++) { accg[mi][ni][j] = 0.f; accu[mi][ni][j] = 0.f; }

    for (int k0 = 0; k0 < HID; k0 += BK) {
        // stage A tile (async direct-to-LDS when available; A is already bf16)
#if HAVE_ASYNC_LDS
        ASYNC_LDS_B128(aSrcBase + k0, aDst0);
        ASYNC_LDS_B128(aSrcBase + k0 + 8, aDst1);
#else
        {
            const uint4* s = (const uint4*)(aSrcBase + k0);
            uint4 d0 = s[0], d1 = s[1];
            *(uint4*)aDst0 = d0;
            *(uint4*)aDst1 = d1;
        }
#endif
        {   // stage B tiles with fp32->bf16 convert
            const float4* sg = (const float4*)(bgSrcBase + k0);
            float4 f0 = sg[0], f1 = sg[1];
            uint4 o;
            o.x = pack2_bf16(f0.x, f0.y); o.y = pack2_bf16(f0.z, f0.w);
            o.z = pack2_bf16(f1.x, f1.y); o.w = pack2_bf16(f1.z, f1.w);
            *(uint4*)&Bg[brr * LDB + bcol] = o;
            const float4* su = (const float4*)(buSrcBase + k0);
            f0 = su[0]; f1 = su[1];
            o.x = pack2_bf16(f0.x, f0.y); o.y = pack2_bf16(f0.z, f0.w);
            o.z = pack2_bf16(f1.x, f1.y); o.w = pack2_bf16(f1.z, f1.w);
            *(uint4*)&Bu[brr * LDB + bcol] = o;
        }
#if HAVE_ASYNC_LDS
        WAIT_ASYNC();
#endif
        __syncthreads();

        BF16Frag a[2], bg[2], bu[2];
#pragma unroll
        for (int mi = 0; mi < 2; mi++) {
            int r = Moff + mi * 16 + lr;
            a[mi].q[0] = *(const uint4*)&As[r * LDA + 16 * hi];       // K = 16*hi..+7
            a[mi].q[1] = *(const uint4*)&As[r * LDA + 16 * hi + 8];   // K = 16*hi+8..+15
        }
#pragma unroll
        for (int ni = 0; ni < 2; ni++) {
            int r = Noff + ni * 16 + lr;
            bg[ni].q[0] = *(const uint4*)&Bg[r * LDB + 16 * hi];      // K = 16*hi..+15
            bg[ni].q[1] = *(const uint4*)&Bg[r * LDB + 16 * hi + 8];
            bu[ni].q[0] = *(const uint4*)&Bu[r * LDB + 16 * hi];
            bu[ni].q[1] = *(const uint4*)&Bu[r * LDB + 16 * hi + 8];
        }
#pragma unroll
        for (int mi = 0; mi < 2; mi++)
#pragma unroll
            for (int ni = 0; ni < 2; ni++) {
                accg[mi][ni] = __builtin_amdgcn_wmma_f32_16x16x32_bf16(
                    false, a[mi].v, false, bg[ni].v, (short)0, accg[mi][ni], false, false);
                accu[mi][ni] = __builtin_amdgcn_wmma_f32_16x16x32_bf16(
                    false, a[mi].v, false, bu[ni].v, (short)0, accu[mi][ni], false, false);
            }
        __syncthreads();
    }

    // epilogue: bias, silu(g)*u, store compacted bf16 activations
#pragma unroll
    for (int ni = 0; ni < 2; ni++) {
        int col = ncol0 + Noff + ni * 16 + lr;          // [0, INTER)
        float bgv = be[col];
        float buv = be[INTER + col];
#pragma unroll
        for (int mi = 0; mi < 2; mi++) {
#pragma unroll
            for (int j = 0; j < 8; j++) {
                int rl = Moff + mi * 16 + 8 * hi + j;   // D: M = j + 8*hi
                int grow = row0 + rl;
                if (grow < n_rows) {
                    float g = accg[mi][ni][j] + bgv;
                    float u = accu[mi][ni][j] + buv;
                    float sv = g * fast_sigmoid(g);
                    act[(size_t)(sbase + grow) * INTER + col] = f32_to_bf16(sv * u);
                }
            }
        }
    }
}

// ---------------------------------------------------------------- GEMM2: act @ w2^T -> out
__global__ __launch_bounds__(256) void gemm2_kernel(
    const unsigned short* __restrict__ act,  // [rows][INTER] bf16 (compacted)
    const float* __restrict__ w,             // [E][H][I] or [H][I]
    const int* __restrict__ base, const int* __restrict__ cnt,
    const int* __restrict__ tok_of, const float* __restrict__ coef_of,
    float* __restrict__ out,                 // [TOKENS][HID] fp32
    int routed) {
    constexpr int BM = 128, BN = 64, BK = 32, LDA = 40, LDB = 40;
    __shared__ __align__(16) unsigned short As[BM * LDA];
    __shared__ __align__(16) unsigned short Bs[BN * LDB];

    const int e      = blockIdx.z;
    const int n_rows = routed ? cnt[e] : TOKENS;
    const int row0   = blockIdx.y * BM;
    if (row0 >= n_rows) return;
    const int sbase  = routed ? base[e] : 0;
    const float* we  = w + (routed ? (size_t)e * HID * INTER : 0);
    const int ncol0  = blockIdx.x * BN;     // output column base in [0, HID)

    const int tid  = threadIdx.x;
    const int lane = tid & 31, wid = tid >> 5;
    const int lr   = lane & 15, hi = lane >> 4;
    const int Moff = (wid & 3) * 32;
    const int Noff = (wid >> 2) * 32;

    const int ar = tid >> 1;
    const int acol = (tid & 1) * 16;
    int arow = row0 + ar;
    if (arow >= n_rows) arow = n_rows - 1;  // clamp within this expert's region
    const unsigned short* aSrcBase = act + (size_t)(sbase + arow) * INTER + acol;
    unsigned short* aDst0 = &As[ar * LDA + acol];
    unsigned short* aDst1 = &As[ar * LDA + acol + 8];

    const int brr = tid >> 2;
    const int bcol = (tid & 3) * 8;
    const float* bSrcBase = we + (size_t)(ncol0 + brr) * INTER + bcol;

    v8f acc[2][2];
#pragma unroll
    for (int mi = 0; mi < 2; mi++)
#pragma unroll
        for (int ni = 0; ni < 2; ni++)
#pragma unroll
            for (int j = 0; j < 8; j++) acc[mi][ni][j] = 0.f;

    for (int k0 = 0; k0 < INTER; k0 += BK) {
#if HAVE_ASYNC_LDS
        ASYNC_LDS_B128(aSrcBase + k0, aDst0);
        ASYNC_LDS_B128(aSrcBase + k0 + 8, aDst1);
#else
        {
            const uint4* s = (const uint4*)(aSrcBase + k0);
            uint4 d0 = s[0], d1 = s[1];
            *(uint4*)aDst0 = d0;
            *(uint4*)aDst1 = d1;
        }
#endif
        {
            const float4* sb = (const float4*)(bSrcBase + k0);
            float4 f0 = sb[0], f1 = sb[1];
            uint4 o;
            o.x = pack2_bf16(f0.x, f0.y); o.y = pack2_bf16(f0.z, f0.w);
            o.z = pack2_bf16(f1.x, f1.y); o.w = pack2_bf16(f1.z, f1.w);
            *(uint4*)&Bs[brr * LDB + bcol] = o;
        }
#if HAVE_ASYNC_LDS
        WAIT_ASYNC();
#endif
        __syncthreads();

        BF16Frag a[2], b[2];
#pragma unroll
        for (int mi = 0; mi < 2; mi++) {
            int r = Moff + mi * 16 + lr;
            a[mi].q[0] = *(const uint4*)&As[r * LDA + 16 * hi];
            a[mi].q[1] = *(const uint4*)&As[r * LDA + 16 * hi + 8];
        }
#pragma unroll
        for (int ni = 0; ni < 2; ni++) {
            int r = Noff + ni * 16 + lr;
            b[ni].q[0] = *(const uint4*)&Bs[r * LDB + 16 * hi];
            b[ni].q[1] = *(const uint4*)&Bs[r * LDB + 16 * hi + 8];
        }
#pragma unroll
        for (int mi = 0; mi < 2; mi++)
#pragma unroll
            for (int ni = 0; ni < 2; ni++)
                acc[mi][ni] = __builtin_amdgcn_wmma_f32_16x16x32_bf16(
                    false, a[mi].v, false, b[ni].v, (short)0, acc[mi][ni], false, false);
        __syncthreads();
    }

#pragma unroll
    for (int ni = 0; ni < 2; ni++) {
        int col = ncol0 + Noff + ni * 16 + lr;
#pragma unroll
        for (int mi = 0; mi < 2; mi++) {
#pragma unroll
            for (int j = 0; j < 8; j++) {
                int rl = Moff + mi * 16 + 8 * hi + j;
                int grow = row0 + rl;
                if (grow < n_rows) {
                    float v = acc[mi][ni][j];
                    if (routed) {
                        int slot = sbase + grow;
                        int t = tok_of[slot];
                        float cf = coef_of[slot];
                        __hip_atomic_fetch_add(&out[(size_t)t * HID + col], cf * v,
                                               __ATOMIC_RELAXED, __HIP_MEMORY_SCOPE_AGENT);
                    } else {
                        out[(size_t)grow * HID + col] = v;  // shared expert writes out first
                    }
                }
            }
        }
    }
}

// ---------------------------------------------------------------- launch
extern "C" void kernel_launch(void* const* d_in, const int* in_sizes, int n_in,
                              void* d_out, int out_size, void* d_ws, size_t ws_size,
                              hipStream_t stream) {
    const float* x   = (const float*)d_in[0];
    const float* gw  = (const float*)d_in[1];
    const float* w1  = (const float*)d_in[2];
    const float* b1  = (const float*)d_in[3];
    const float* w2  = (const float*)d_in[4];
    const float* sgw = (const float*)d_in[5];
    const float* sgb = (const float*)d_in[6];
    const float* sdw = (const float*)d_in[7];
    float* out = (float*)d_out;

    char* ws = (char*)d_ws;
    size_t off = 0;
    auto alloc = [&](size_t bytes) -> char* {
        char* p = ws + off;
        off += (bytes + 255) & ~(size_t)255;
        return p;
    };
    unsigned short* xb    = (unsigned short*)alloc((size_t)TOKENS * HID * 2);      // 16 MB
    unsigned short* act_s = (unsigned short*)alloc((size_t)TOKENS * INTER * 2);    //  8 MB
    unsigned short* act_r = (unsigned short*)alloc((size_t)2 * TOKENS * INTER * 2);// 16 MB
    int*   cntcur  = (int*)  alloc(32 * sizeof(int));   // [0:16) cnt, [16:32) cursor
    int*   basep   = (int*)  alloc(16 * sizeof(int));
    int*   pick_e  = (int*)  alloc(2 * TOKENS * sizeof(int));
    float* pick_w  = (float*)alloc(2 * TOKENS * sizeof(float));
    int*   tok_of  = (int*)  alloc(2 * TOKENS * sizeof(int));
    float* coef_of = (float*)alloc(2 * TOKENS * sizeof(float));
    int* cnt = cntcur;
    int* cursor = cntcur + 16;
    (void)in_sizes; (void)n_in; (void)out_size; (void)ws_size;

    (void)hipMemsetAsync(cntcur, 0, 32 * sizeof(int), stream);

    cvt_x_kernel<<<(TOKENS * HID / 8) / 256, 256, 0, stream>>>(x, (uint4*)xb);
    router_kernel<<<TOKENS / 256, 256, 0, stream>>>(x, gw, cnt, pick_e, pick_w);
    scan_kernel<<<1, 32, 0, stream>>>(cnt, basep);
    assign_kernel<<<TOKENS / 256, 256, 0, stream>>>(pick_e, pick_w, basep, cursor, tok_of, coef_of);

    // shared expert (writes out densely first)
    gemm1_kernel<<<dim3(INTER / 64, TOKENS / 128, 1), 256, 0, stream>>>(
        xb, sgw, sgb, basep, cnt, tok_of, act_s, 0);
    gemm2_kernel<<<dim3(HID / 64, TOKENS / 128, 1), 256, 0, stream>>>(
        act_s, sdw, basep, cnt, tok_of, coef_of, out, 0);

    // routed experts (scatter-accumulate into out)
    gemm1_kernel<<<dim3(INTER / 64, TOKENS / 128, NEXP), 256, 0, stream>>>(
        xb, w1, b1, basep, cnt, tok_of, act_r, 1);
    gemm2_kernel<<<dim3(HID / 64, TOKENS / 128, NEXP), 256, 0, stream>>>(
        act_r, w2, basep, cnt, tok_of, coef_of, out, 1);
}
